// GPT_87179246174369
// MI455X (gfx1250) — matrix operand order repaired and checked by reading, not compiled, not run on hardware
//
#include <hip/hip_runtime.h>
#include <math.h>

// GPT-2 small forward, CDNA5 (gfx1250): wave32, WMMA f16->f32 everywhere,
// TDM (tensor_load_to_lds) GEMM staging, async-to-LDS attention staging,
// WMMA flash-attention.
#define BB   4
#define TT   1024
#define LL   12
#define HH   12
#define DD   768
#define HDIM 64
#define FFD  3072
#define VV   50257
#define MT   (BB*TT)      // 4096 rows of activations

#define USE_TDM   1       // tensor_load_to_lds B staging in GEMM
#define USE_ASYNC 1       // global_load_async_to_lds_b128 staging (attention / fallback)

typedef _Float16 f16;
typedef __attribute__((ext_vector_type(16))) _Float16 v16h;
typedef __attribute__((ext_vector_type(8)))  _Float16 v8h;
typedef __attribute__((ext_vector_type(8)))  float    v8f;
typedef __attribute__((ext_vector_type(4)))  unsigned v4u;
typedef __attribute__((ext_vector_type(8)))  int      v8i;
typedef __attribute__((ext_vector_type(4)))  int      v4i;

__device__ __forceinline__ float gelu_f(float x) {
  const float c = 0.7978845608028654f; // sqrt(2/pi)
  return 0.5f * x * (1.0f + tanhf(c * (x + 0.044715f * x * x * x)));
}

__device__ __forceinline__ v16h halves_cat(v8h a, v8h b) {
  return __builtin_shufflevector(a, b, 0,1,2,3,4,5,6,7,8,9,10,11,12,13,14,15);
}

__device__ __forceinline__ v8f wmma_f16(v16h a, v16h b, v8f c) {
  return __builtin_amdgcn_wmma_f32_16x16x32_f16(false, a, false, b, (short)0, c,
                                                false, false);
}

// Build a 16x32 A-fragment for one lane from two contiguous 8-float runs
// (contiguous k = 16*hi + s at call sites: a[0..7] = p0[0..7], a[8..15] = p1[0..7]), scaling on convert.
__device__ __forceinline__ v16h load_a_frag(const float* p0, const float* p1,
                                            float scale) {
  const float4* a0 = (const float4*)p0;
  const float4* a1 = (const float4*)p1;
  float4 x0 = a0[0], x1 = a0[1];
  float4 x2 = a1[0], x3 = a1[1];
  v16h a;
  a[0]  = (f16)(x0.x*scale); a[1]  = (f16)(x0.y*scale);
  a[2]  = (f16)(x0.z*scale); a[3]  = (f16)(x0.w*scale);
  a[4]  = (f16)(x1.x*scale); a[5]  = (f16)(x1.y*scale);
  a[6]  = (f16)(x1.z*scale); a[7]  = (f16)(x1.w*scale);
  a[8]  = (f16)(x2.x*scale); a[9]  = (f16)(x2.y*scale);
  a[10] = (f16)(x2.z*scale); a[11] = (f16)(x2.w*scale);
  a[12] = (f16)(x3.x*scale); a[13] = (f16)(x3.y*scale);
  a[14] = (f16)(x3.z*scale); a[15] = (f16)(x3.w*scale);
  return a;
}

// ---------------------------------------------------------------------------
// Embedding: x[bt, d] = wte[idx[bt], d] + wpe[t, d]
// ---------------------------------------------------------------------------
__global__ __launch_bounds__(256) void embed_kernel(
    const int* __restrict__ idx, const float* __restrict__ wte,
    const float* __restrict__ wpe, float* __restrict__ x)
{
  int i = blockIdx.x * 256 + threadIdx.x;
  if (i >= MT * DD) return;
  int d  = i % DD;
  int bt = i / DD;
  int t  = bt % TT;
  x[i] = wte[(size_t)idx[bt] * DD + d] + wpe[(size_t)t * DD + d];
}

// ---------------------------------------------------------------------------
// LayerNorm: one 256-thread block per row of D=768
// ---------------------------------------------------------------------------
__global__ __launch_bounds__(256) void ln_kernel(
    const float* __restrict__ x, const float* __restrict__ w,
    const float* __restrict__ b, float* __restrict__ y)
{
  __shared__ float red[256];
  const int row = blockIdx.x;
  const float* xr = x + (size_t)row * DD;

  float s = 0.f;
  for (int i = threadIdx.x; i < DD; i += 256) s += xr[i];
  red[threadIdx.x] = s; __syncthreads();
  for (int off = 128; off > 0; off >>= 1) {
    if (threadIdx.x < off) red[threadIdx.x] += red[threadIdx.x + off];
    __syncthreads();
  }
  float mean = red[0] * (1.0f / DD);
  __syncthreads();

  float v = 0.f;
  for (int i = threadIdx.x; i < DD; i += 256) { float d = xr[i] - mean; v += d * d; }
  red[threadIdx.x] = v; __syncthreads();
  for (int off = 128; off > 0; off >>= 1) {
    if (threadIdx.x < off) red[threadIdx.x] += red[threadIdx.x + off];
    __syncthreads();
  }
  float rstd = rsqrtf(red[0] * (1.0f / DD) + 1e-5f);

  float* yr = y + (size_t)row * DD;
  for (int i = threadIdx.x; i < DD; i += 256)
    yr[i] = (xr[i] - mean) * rstd * w[i] + b[i];
}

// ---------------------------------------------------------------------------
// GEMM via V_WMMA_F32_16X16X32_F16 with LDS-staged B tiles.
// C[M,N] = act(A[M,K] @ B + bias) (+ residual)
// BT=0: B row-major [K,N], requires N % 64 == 0. B tile (32x64 f32) staged by
//       the Tensor Data Mover (one D# per K-step, wave 0 issues, TENSORcnt),
//       then converted to f16 cooperatively. Double-buffered.
// BT=1: B row-major [N,K] (A @ B^T), ragged N allowed (direct staging).
// Block: 256 threads = 8 waves; block tile 128x64; B tile staged once, shared.
// ---------------------------------------------------------------------------
template<int BT>
__global__ __launch_bounds__(256) void gemm_wmma_kernel(
    const float* __restrict__ A, const float* __restrict__ Bm,
    const float* __restrict__ bias, const float* __restrict__ res,
    float* __restrict__ C, int M, int N, int K, int doGelu)
{
  __shared__ f16 bsh[2][64][40];        // f16 B tiles, [n][k], padded rows
#if USE_TDM || USE_ASYNC
  __shared__ float bstage[2][32][64];   // raw f32 landing tiles, [k][n]
#endif

  const int tid  = threadIdx.x;
  const int lane = tid & 31;
  const int wave = tid >> 5;
  const int lr   = lane & 15;
  const int hi   = lane >> 4;
  const int m0   = blockIdx.y * 128 + wave * 16;
  const int n0   = blockIdx.x * 64;
  const int ksteps = K >> 5;

  const v8f zero = {0.f,0.f,0.f,0.f,0.f,0.f,0.f,0.f};
  v8f acc[4] = {zero, zero, zero, zero};

  const float* arow = A + (size_t)(m0 + lr) * K;

  auto stage_direct = [&](int buf, int kb) {
    if (BT) {
      int n = tid >> 2, kh = tid & 3;
      int col = n0 + n;
      int nc  = col < N ? col : (N - 1);
      const float4* gp = (const float4*)(Bm + (size_t)nc * K + kb + kh * 8);
      float4 x0 = gp[0], x1 = gp[1];
      f16* dst = &bsh[buf][n][kh * 8];
      dst[0] = (f16)x0.x; dst[1] = (f16)x0.y; dst[2] = (f16)x0.z; dst[3] = (f16)x0.w;
      dst[4] = (f16)x1.x; dst[5] = (f16)x1.y; dst[6] = (f16)x1.z; dst[7] = (f16)x1.w;
    } else {
      int n = tid & 63, kh = tid >> 6;
      int col = n0 + n;
      int nc  = col < N ? col : (N - 1);
      f16* dst = &bsh[buf][n][kh * 8];
#pragma unroll
      for (int j = 0; j < 8; ++j)
        dst[j] = (f16)Bm[(size_t)(kb + kh * 8 + j) * N + nc];
    }
  };

#if USE_TDM || USE_ASYNC
  // Cooperative f32 -> f16 transpose bstage -> bsh
  auto stage_convert = [&](int buf) {
    int n = tid & 63, kh = tid >> 6;
    f16* dst = &bsh[buf][n][kh * 8];
#pragma unroll
    for (int j = 0; j < 8; ++j)
      dst[j] = (f16)bstage[buf][kh * 8 + j][n];
  };
#endif

#if USE_TDM
  // One TDM descriptor moves the whole 32x64 f32 tile (row stride N) to LDS.
  auto tdm_stage = [&](int buf, int kb) {
    unsigned long long ga = (unsigned long long)(Bm + (size_t)kb * N + n0);
    unsigned lds = (unsigned)(size_t)&bstage[buf][0][0];
    // D# group 0: count=1 | lds_addr | global_addr[56:0] | type=2
    v4u g0 = { 1u, lds, (unsigned)ga,
               (unsigned)((ga >> 32) & 0x1ffffffull) | (2u << 30) };
    // D# group 1: data_size=4B; tensor_dim0=64; tensor_dim1=32;
    //             tile_dim0=64; tile_dim1=32; tensor_dim0_stride=N
    v8i g1 = { (int)(2u << 16),
               (int)(64u << 16),
               (int)(32u << 16),
               (int)(64u << 16),
               32,
               N,
               0, 0 };
    v4i g2 = {0,0,0,0};
    v4i g3 = {0,0,0,0};
    v8i g4 = {0,0,0,0,0,0,0,0};
    // amdgpu-toolchain (clang-23) 6-arg form
    __builtin_amdgcn_tensor_load_to_lds(g0, g1, g2, g3, g4, 0);
  };
#elif USE_ASYNC
  auto stage_async_copy = [&](int buf, int kb) {
#pragma unroll
    for (int j = 0; j < 2; ++j) {
      int cid = tid + 256 * j;        // 512 x 16B chunks = 8KB tile
      int row = cid >> 4;
      int cof = (cid & 15) << 2;
      const float* gp = Bm + (size_t)(kb + row) * N + n0 + cof;
      unsigned la = (unsigned)(size_t)&bstage[buf][row][cof];
      unsigned long long ga = (unsigned long long)gp;
      asm volatile("global_load_async_to_lds_b128 %0, %1, off"
                   :: "v"(la), "v"(ga) : "memory");
    }
  };
#endif

  auto compute = [&](int buf, int k0) {
    __builtin_prefetch(arow + k0 + 64, 0, 1);   // global_prefetch_b8
    v16h a = load_a_frag(arow + k0 + 16 * hi, arow + k0 + 16 * hi + 8, 1.0f);
#pragma unroll
    for (int t = 0; t < 4; ++t) {
      const v8h* bp = (const v8h*)&bsh[buf][t * 16 + lr][hi * 16];
      v16h bfrag = halves_cat(bp[0], bp[1]);
      acc[t] = wmma_f16(a, bfrag, acc[t]);
    }
  };

  if (!BT) {
#if USE_TDM
    const bool w0 = (__builtin_amdgcn_readfirstlane(wave) == 0);  // scalar branch
    if (w0) tdm_stage(0, 0);
    for (int step = 0; step < ksteps; ++step) {
      int buf = step & 1, k0 = step << 5;
      if (w0) __builtin_amdgcn_s_wait_tensorcnt(0);
      __syncthreads();                  // raw tile landed for everyone
      stage_convert(buf);
      __syncthreads();                  // f16 tile ready
      if (w0 && step + 1 < ksteps) tdm_stage(buf ^ 1, k0 + 32);
      compute(buf, k0);
    }
#elif USE_ASYNC
    stage_async_copy(0, 0);
    for (int step = 0; step < ksteps; ++step) {
      int buf = step & 1, k0 = step << 5;
      asm volatile("s_wait_asynccnt 0" ::: "memory");
      __syncthreads();
      stage_convert(buf);
      __syncthreads();
      if (step + 1 < ksteps) stage_async_copy(buf ^ 1, k0 + 32);
      compute(buf, k0);
    }
#else
    stage_direct(0, 0);
    for (int step = 0; step < ksteps; ++step) {
      int buf = step & 1, k0 = step << 5;
      __syncthreads();
      if (step + 1 < ksteps) stage_direct(buf ^ 1, k0 + 32);
      compute(buf, k0);
    }
#endif
  } else {
    stage_direct(0, 0);
    for (int step = 0; step < ksteps; ++step) {
      int buf = step & 1, k0 = step << 5;
      __syncthreads();
      if (step + 1 < ksteps) stage_direct(buf ^ 1, k0 + 32);
      compute(buf, k0);
    }
  }

  // Epilogue. C/D layout: VGPR r -> row m0 + r + 8*hi, col = n0 + 16*t + lr.
#pragma unroll
  for (int t = 0; t < 4; ++t) {
    int col = n0 + t * 16 + lr;
    if (col < N) {
#pragma unroll
      for (int r = 0; r < 8; ++r) {
        int row = m0 + r + hi * 8;
        float v = acc[t][r];
        if (bias)   v += bias[col];
        if (doGelu) v  = gelu_f(v);
        if (res)    v += res[(size_t)row * N + col];
        C[(size_t)row * N + col] = v;
      }
    }
  }
}

// ---------------------------------------------------------------------------
// WMMA flash attention. Grid: (T/128, B*H); block = 256 threads = 8 waves.
// Wave w owns 16 query rows. Per 32-key step: K/V blocks staged via
// global_load_async_to_lds (raw f32), converted to f16 fragment layouts,
// then S = Q K^T (4 WMMAs), online softmax, Y += P V (4 WMMAs).
// qkv rows: [q(768) | k(768) | v(768)], head h at cols h*64..h*64+63.
// ---------------------------------------------------------------------------
__global__ __launch_bounds__(256) void attn_wmma_kernel(
    const float* __restrict__ qkv, float* __restrict__ y)
{
  __shared__ f16 ksh[32][72];       // [key][head-dim], padded rows
  __shared__ f16 vsht[64][40];      // [head-dim][key] (V transposed), padded
  __shared__ f16 psh[8][16][40];    // per-wave P scratch, [row][key], padded
#if USE_ASYNC
  __shared__ float kstage[32][64];  // raw f32 async landing tiles
  __shared__ float vstage[32][64];
#endif

  const int tid  = threadIdx.x;
  const int lane = tid & 31;
  const int wave = tid >> 5;
  const int lr   = lane & 15;
  const int hi   = lane >> 4;
  const int qt   = blockIdx.x;
  const int h    = blockIdx.y % HH;
  const int b    = blockIdx.y / HH;
  const int m0a  = qt * 128 + wave * 16;   // absolute query row in [0,T)
  const int m0a_u = __builtin_amdgcn_readfirstlane(m0a);  // provably uniform

  const size_t rstr = 3 * DD;
  const float* qbase = qkv + (size_t)b * TT * rstr + h * HDIM;
  const float* kbase = qbase + DD;
  const float* vbase = qbase + 2 * DD;

  // Q fragments (rows m0a+lr), pre-scaled by 1/sqrt(HD)
  const float* qrow = qbase + (size_t)(m0a + lr) * rstr;
  v16h qa0 = load_a_frag(qrow +      16 * hi, qrow +  8 + 16 * hi, 0.125f);
  v16h qa1 = load_a_frag(qrow + 32 + 16 * hi, qrow + 40 + 16 * hi, 0.125f);

  const v8f zero = {0.f,0.f,0.f,0.f,0.f,0.f,0.f,0.f};
  v8f acc[4] = {zero, zero, zero, zero};
  float m_r[8], l_r[8];
#pragma unroll
  for (int r = 0; r < 8; ++r) { m_r[r] = -3.0e38f; l_r[r] = 0.f; }

#if USE_ASYNC
  auto stage_copy = [&](int s0) {     // issue async copies for K and V blocks
#pragma unroll
    for (int j = 0; j < 2; ++j) {
      int cid = tid + 256 * j;        // 512 x 16B chunks per matrix
      int key = cid >> 4;
      int cof = (cid & 15) << 2;
      unsigned long long gk =
          (unsigned long long)(kbase + (size_t)(s0 + key) * rstr + cof);
      unsigned lk = (unsigned)(size_t)&kstage[key][cof];
      asm volatile("global_load_async_to_lds_b128 %0, %1, off"
                   :: "v"(lk), "v"(gk) : "memory");
      unsigned long long gv =
          (unsigned long long)(vbase + (size_t)(s0 + key) * rstr + cof);
      unsigned lv = (unsigned)(size_t)&vstage[key][cof];
      asm volatile("global_load_async_to_lds_b128 %0, %1, off"
                   :: "v"(lv), "v"(gv) : "memory");
    }
  };
  auto stage_convert = [&]() {        // kstage/vstage -> ksh / vsht (f16)
    int key = tid >> 3, koff = (tid & 7) * 8;
    f16* kd = &ksh[key][koff];
#pragma unroll
    for (int j = 0; j < 8; ++j) kd[j] = (f16)kstage[key][koff + j];
#pragma unroll
    for (int j = 0; j < 8; ++j) vsht[koff + j][key] = (f16)vstage[key][koff + j];
  };
#endif

  const int nsb = 4 * (qt + 1);            // 32-key blocks to cover diagonal
#if USE_ASYNC
  stage_copy(0);
#endif
  for (int sbi = 0; sbi < nsb; ++sbi) {
    const int s0 = sbi * 32;
#if USE_ASYNC
    asm volatile("s_wait_asynccnt 0" ::: "memory");
    __syncthreads();                        // raw tiles landed; prev reads done
    stage_convert();
    __syncthreads();                        // f16 tiles visible block-wide
    if (sbi + 1 < nsb) stage_copy(s0 + 32); // overlap next copy with compute
#else
    __syncthreads();
    {
      int key = tid >> 3, koff = (tid & 7) * 8;
      const float4* kp = (const float4*)(kbase + (size_t)(s0 + key) * rstr + koff);
      float4 k0 = kp[0], k1 = kp[1];
      f16* kd = &ksh[key][koff];
      kd[0] = (f16)k0.x; kd[1] = (f16)k0.y; kd[2] = (f16)k0.z; kd[3] = (f16)k0.w;
      kd[4] = (f16)k1.x; kd[5] = (f16)k1.y; kd[6] = (f16)k1.z; kd[7] = (f16)k1.w;
      const float4* vp = (const float4*)(vbase + (size_t)(s0 + key) * rstr + koff);
      float4 v0 = vp[0], v1 = vp[1];
      vsht[koff + 0][key] = (f16)v0.x; vsht[koff + 1][key] = (f16)v0.y;
      vsht[koff + 2][key] = (f16)v0.z; vsht[koff + 3][key] = (f16)v0.w;
      vsht[koff + 4][key] = (f16)v1.x; vsht[koff + 5][key] = (f16)v1.y;
      vsht[koff + 6][key] = (f16)v1.z; vsht[koff + 7][key] = (f16)v1.w;
    }
    __syncthreads();
#endif

    if (s0 <= m0a_u + 15) {                 // scalar (s_cbranch), EXEC stays full
      // S tiles: keys [s0, s0+16) and [s0+16, s0+32)
      v8f st0 = zero, st1 = zero;
#pragma unroll
      for (int ks = 0; ks < 2; ++ks) {
        v16h qa = ks ? qa1 : qa0;
        const v8h* bp0 = (const v8h*)&ksh[lr][ks * 32 + 16 * hi];
        st0 = wmma_f16(qa, halves_cat(bp0[0], bp0[1]), st0);
        const v8h* bp1 = (const v8h*)&ksh[16 + lr][ks * 32 + 16 * hi];
        st1 = wmma_f16(qa, halves_cat(bp1[0], bp1[1]), st1);
      }

      // Causal mask (only near the diagonal)
      if (s0 + 31 > m0a_u) {
        int key0 = s0 + lr, key1 = s0 + 16 + lr;
#pragma unroll
        for (int r = 0; r < 8; ++r) {
          int rowq = m0a + r + 8 * hi;
          if (key0 > rowq) st0[r] = -3.0e38f;
          if (key1 > rowq) st1[r] = -3.0e38f;
        }
      }

      // Online softmax update; row stats reduced across the 16-lane half
      float mx[8], p0[8], p1[8], rs[8], cr[8];
#pragma unroll
      for (int r = 0; r < 8; ++r) mx[r] = fmaxf(st0[r], st1[r]);
#pragma unroll
      for (int msk = 1; msk <= 8; msk <<= 1)
#pragma unroll
        for (int r = 0; r < 8; ++r) mx[r] = fmaxf(mx[r], __shfl_xor(mx[r], msk));
#pragma unroll
      for (int r = 0; r < 8; ++r) {
        float nm = fmaxf(m_r[r], mx[r]);
        cr[r] = __expf(m_r[r] - nm);
        p0[r] = __expf(st0[r] - nm);
        p1[r] = __expf(st1[r] - nm);
        rs[r] = p0[r] + p1[r];
        m_r[r] = nm;
      }
#pragma unroll
      for (int msk = 1; msk <= 8; msk <<= 1)
#pragma unroll
        for (int r = 0; r < 8; ++r) rs[r] += __shfl_xor(rs[r], msk);
#pragma unroll
      for (int r = 0; r < 8; ++r) l_r[r] = l_r[r] * cr[r] + rs[r];
#pragma unroll
      for (int t = 0; t < 4; ++t)
#pragma unroll
        for (int r = 0; r < 8; ++r) acc[t][r] *= cr[r];

      // P (C-layout) -> per-wave LDS -> A-fragment layout
#pragma unroll
      for (int r = 0; r < 8; ++r) {
        psh[wave][r + 8 * hi][lr]      = (f16)p0[r];
        psh[wave][r + 8 * hi][16 + lr] = (f16)p1[r];
      }
      asm volatile("s_wait_dscnt 0" ::: "memory");  // wave-local RAW on LDS
      const v8h* pa0 = (const v8h*)&psh[wave][lr][16 * hi];
      const v8h* pa1 = (const v8h*)&psh[wave][lr][16 * hi + 8];
      v16h pa = halves_cat(pa0[0], pa1[0]);

      // Y += P V
#pragma unroll
      for (int t = 0; t < 4; ++t) {
        const v8h* vp2 = (const v8h*)&vsht[t * 16 + lr][16 * hi];
        acc[t] = wmma_f16(pa, halves_cat(vp2[0], vp2[1]), acc[t]);
      }
    }
  }

  // Write normalized output
#pragma unroll
  for (int t = 0; t < 4; ++t) {
    int dim = t * 16 + lr;
#pragma unroll
    for (int r = 0; r < 8; ++r) {
      int rowq = m0a + r + 8 * hi;
      y[((size_t)(b * TT + rowq)) * DD + h * HDIM + dim] = acc[t][r] / l_r[r];
    }
  }
}

// ---------------------------------------------------------------------------
// Host orchestration
// ---------------------------------------------------------------------------
static inline void run_gemm(const float* A, const float* Bmat, const float* bias,
                            const float* res, float* C, int M, int N, int K,
                            bool btrans, int doGelu, hipStream_t stream)
{
  dim3 grid((N + 63) / 64, M / 128);
  if (btrans)
    gemm_wmma_kernel<1><<<grid, 256, 0, stream>>>(A, Bmat, bias, res, C, M, N, K, doGelu);
  else
    gemm_wmma_kernel<0><<<grid, 256, 0, stream>>>(A, Bmat, bias, res, C, M, N, K, doGelu);
}

extern "C" void kernel_launch(void* const* d_in, const int* in_sizes, int n_in,
                              void* d_out, int out_size, void* d_ws, size_t ws_size,
                              hipStream_t stream)
{
  (void)in_sizes; (void)n_in; (void)out_size; (void)ws_size;

  const int*   idx    = (const int*)  d_in[0];
  const float* wte    = (const float*)d_in[1];
  const float* wpe    = (const float*)d_in[2];
  const float* ln1_w  = (const float*)d_in[3];
  const float* ln1_b  = (const float*)d_in[4];
  const float* w_qkv  = (const float*)d_in[5];
  const float* b_qkv  = (const float*)d_in[6];
  const float* w_ao   = (const float*)d_in[7];
  const float* b_ao   = (const float*)d_in[8];
  const float* ln2_w  = (const float*)d_in[9];
  const float* ln2_b  = (const float*)d_in[10];
  const float* w_fc   = (const float*)d_in[11];
  const float* b_fc   = (const float*)d_in[12];
  const float* w_mo   = (const float*)d_in[13];
  const float* b_mo   = (const float*)d_in[14];
  const float* lnf_w  = (const float*)d_in[15];
  const float* lnf_b  = (const float*)d_in[16];
  const float* w_head = (const float*)d_in[17];
  float* out = (float*)d_out;

  // Workspace layout (floats)
  float* X   = (float*)d_ws;              // [MT, D]   residual stream
  float* Hb  = X   + (size_t)MT * DD;     // [MT, D]   layernorm output
  float* QKV = Hb  + (size_t)MT * DD;     // [MT, 3D]
  float* Y   = QKV + (size_t)MT * 3 * DD; // [MT, D]   attention output
  float* FC  = Y   + (size_t)MT * DD;     // [MT, FF]

  // Embedding
  embed_kernel<<<(MT * DD + 255) / 256, 256, 0, stream>>>(idx, wte, wpe, X);

  for (int l = 0; l < LL; ++l) {
    const float* lqkv = w_qkv + (size_t)l * DD * 3 * DD;
    const float* lao  = w_ao  + (size_t)l * DD * DD;
    const float* lfc  = w_fc  + (size_t)l * DD * FFD;
    const float* lmo  = w_mo  + (size_t)l * FFD * DD;

    // h = ln1(x)
    ln_kernel<<<MT, 256, 0, stream>>>(X, ln1_w + (size_t)l * DD, ln1_b + (size_t)l * DD, Hb);
    // qkv = h @ w_qkv + b_qkv
    run_gemm(Hb, lqkv, b_qkv + (size_t)l * 3 * DD, nullptr, QKV, MT, 3 * DD, DD, false, 0, stream);
    // y = attention(qkv)
    attn_wmma_kernel<<<dim3(TT / 128, BB * HH), 256, 0, stream>>>(QKV, Y);
    // x = x + y @ w_ao + b_ao   (residual fused, in-place on X)
    run_gemm(Y, lao, b_ao + (size_t)l * DD, X, X, MT, DD, DD, false, 0, stream);
    // h = ln2(x)
    ln_kernel<<<MT, 256, 0, stream>>>(X, ln2_w + (size_t)l * DD, ln2_b + (size_t)l * DD, Hb);
    // fc = gelu(h @ w_fc + b_fc)
    run_gemm(Hb, lfc, b_fc + (size_t)l * FFD, nullptr, FC, MT, FFD, DD, false, 1, stream);
    // x = x + fc @ w_mo + b_mo  (residual fused, in-place on X)
    run_gemm(FC, lmo, b_mo + (size_t)l * DD, X, X, MT, DD, FFD, false, 0, stream);
  }

  // Final layernorm + LM head (A @ w_head^T)
  ln_kernel<<<MT, 256, 0, stream>>>(X, lnf_w, lnf_b, Hb);
  run_gemm(Hb, w_head, nullptr, nullptr, out, MT, VV, DD, true, 0, stream);
}
